// ConditionalRoutedFeedForward_46574625358486
// MI455X (gfx1250) — matrix operand order repaired and checked by reading, not compiled, not run on hardware
//
#include <hip/hip_runtime.h>
#include <math.h>

// ---------------- types ----------------
typedef __attribute__((ext_vector_type(16))) __bf16 v16bf;
typedef __attribute__((ext_vector_type(8)))  float  v8f;

#define DIM   1024
#define NTOK  65536        // b*n = 8*8192
#define NSEQ  8192
#define NB    8
#define KSEL  1024         // NUM_HEAVY
#define TM    32           // token tile per block (2 M-tiles)
#define CH    256          // hidden chunk width
#define XST   516          // Xn row stride in u32 (512 + 4 pad -> bank stagger)
#define HST   264          // Hc row stride in halves (256 + 8 pad)

// ---------------- helpers ----------------
__device__ __forceinline__ unsigned pack2_bf16(float a, float b) {
  unsigned short ua = __builtin_bit_cast(unsigned short, (__bf16)a);
  unsigned short ub = __builtin_bit_cast(unsigned short, (__bf16)b);
  return (unsigned)ua | ((unsigned)ub << 16);
}

__device__ __forceinline__ float gelu_exact(float x) {
  return 0.5f * x * (1.0f + erff(x * 0.70710678118654752f));
}

// A operand (16x32 bf16). Per lane: row m, K = {kb..kb+15}
// (kb already includes the +16 offset for lanes 16-31). Pairs are contiguous
// -> lowers to two ds_load_b128.
__device__ __forceinline__ v16bf load_a_lds(const unsigned* base, int strideU,
                                            int m, int kb) {
  v16bf a;
  const unsigned* rp = base + m * strideU;
#pragma unroll
  for (int p = 0; p < 8; ++p) {
    int K = kb + 2 * p;   // even
    unsigned u = rp[K >> 1];
    a[2 * p]     = __builtin_bit_cast(__bf16, (unsigned short)(u & 0xffffu));
    a[2 * p + 1] = __builtin_bit_cast(__bf16, (unsigned short)(u >> 16));
  }
  return a;
}

// B operand (32x16 bf16): 16 contiguous K halves from one pre-converted
// bf16 weight row; 32B aligned -> two global_load_b128, no cvt.
__device__ __forceinline__ v16bf load_b_bf16(const __bf16* __restrict__ p) {
  return *(const v16bf*)p;
}

// ---------------- weight fp32 -> bf16 pre-conversion ----------------
__global__ __launch_bounds__(256) void convert_kernel(
    const float* __restrict__ src, unsigned* __restrict__ dst, int n) {
  int i = (blockIdx.x * 256 + threadIdx.x) * 8;
  if (i >= n) return;
  float4 a = *(const float4*)(src + i);
  float4 b = *(const float4*)(src + i + 4);
  uint4 o;
  o.x = pack2_bf16(a.x, a.y);
  o.y = pack2_bf16(a.z, a.w);
  o.z = pack2_bf16(b.x, b.y);
  o.w = pack2_bf16(b.z, b.w);
  *(uint4*)(dst + i / 2) = o;
}

// ---------------- router: s[t] = dot(x[t,:], rt) ----------------
__global__ __launch_bounds__(256) void router_kernel(
    const float* __restrict__ x, const float* __restrict__ rt,
    float* __restrict__ s_out) {
  const int lane = threadIdx.x & 31;
  const int t = blockIdx.x * 8 + (threadIdx.x >> 5);
  const float* xr = x + (size_t)t * DIM;
  float sum = 0.f;
#pragma unroll 8
  for (int k = lane; k < DIM; k += 32) sum += xr[k] * rt[k];
#pragma unroll
  for (int o = 16; o > 0; o >>= 1) sum += __shfl_xor(sum, o, 32);
  if (lane == 0) s_out[t] = sum;
}

// ---------------- coor_descent (one block per batch row) ----------------
__global__ __launch_bounds__(1024) void coor_kernel(
    const float* __restrict__ s_in, float* __restrict__ scores) {
  __shared__ float red[1024];
  const int bb = blockIdx.x;
  const int tid = threadIdx.x;
  const float* srow = s_in + (size_t)bb * NSEQ;
  const float logk = logf(1152.0f);  // min(1024*9/8, 8192)

  float sl[8], bl[8];
#pragma unroll
  for (int i = 0; i < 8; ++i) {
    sl[i] = srow[tid + i * 1024];
    bl[i] = -sl[i];
  }
  float a = 0.f;
  for (int it = 0; it < 50; ++it) {
    float m = -3.4e38f;
#pragma unroll
    for (int i = 0; i < 8; ++i) m = fmaxf(m, sl[i] + bl[i]);
    red[tid] = m;
    __syncthreads();
    for (int o = 512; o > 0; o >>= 1) {
      if (tid < o) red[tid] = fmaxf(red[tid], red[tid + o]);
      __syncthreads();
    }
    const float mx = red[0];
    __syncthreads();
    float sum = 0.f;
#pragma unroll
    for (int i = 0; i < 8; ++i) sum += expf(sl[i] + bl[i] - mx);
    red[tid] = sum;
    __syncthreads();
    for (int o = 512; o > 0; o >>= 1) {
      if (tid < o) red[tid] = red[tid] + red[tid + o];
      __syncthreads();
    }
    const float S = red[0];
    __syncthreads();
    a = logk - (mx + logf(S));
#pragma unroll
    for (int i = 0; i < 8; ++i) bl[i] = -fmaxf(sl[i] + a, 0.f);
  }
#pragma unroll
  for (int i = 0; i < 8; ++i)
    scores[(size_t)bb * NSEQ + tid + i * 1024] = expf(fminf(sl[i] + a, 0.f));
}

// ---------------- top-k via stable rank (matches lax.top_k ties) --------
__global__ __launch_bounds__(256) void rank_kernel(
    const float* __restrict__ scores, int* __restrict__ sel) {
  __shared__ float sc[NSEQ];
  const int bb = blockIdx.x >> 5;
  const int blk = blockIdx.x & 31;
  const float* srow = scores + (size_t)bb * NSEQ;
  for (int j = threadIdx.x; j < NSEQ; j += 256) sc[j] = srow[j];
  __syncthreads();
  const int i = blk * 256 + threadIdx.x;
  const float v = sc[i];
  int cnt = 0;
  for (int j = 0; j < NSEQ; ++j) {
    float u = sc[j];
    cnt += (u > v) || (u == v && j < i);
  }
  if (cnt < KSEL) sel[bb * KSEL + cnt] = i;
}

// ---------------- fused FFN (rmsnorm -> GEMM1 -> gelu -> GEMM2) ---------
// HEAVY: gather rows through sel[] and += into out; else direct write.
// 32-token tile, 8 waves; wave w: M-tile = w&1, N-group = w>>1.
template <int DH, int NCHUNK, bool HEAVY>
__global__ __launch_bounds__(256) void ffn_kernel(
    const float* __restrict__ X, const float* __restrict__ gamma,
    const __bf16* __restrict__ W1, const float* __restrict__ B1,
    const __bf16* __restrict__ W2, const float* __restrict__ B2,
    const int* __restrict__ sel, float* __restrict__ out) {
  __shared__ unsigned XnU[TM * XST];            // TM x DIM bf16, padded rows
  __shared__ __bf16 Hcf[TM * HST];              // gelu'd hidden chunk, padded
  __shared__ float rr[TM][9];
  __shared__ unsigned long long rowOff[TM];

  const int tid = threadIdx.x;
  const int lane = tid & 31;
  const int w = tid >> 5;
  const int m0 = blockIdx.x * TM;
  const int mlow = lane & 15;
  const int khalf = lane >> 4;
  const int mtile = w & 1;
  const int ngrp = w >> 1;                      // 0..3
  const int mbase = mtile * 16;

  if (tid < TM) {
    unsigned long long tok;
    if (HEAVY) {
      int bb = (m0 + tid) >> 10;
      int slot = (m0 + tid) & 1023;
      tok = (unsigned long long)bb * NSEQ + (unsigned)sel[bb * KSEL + slot];
    } else {
      tok = (unsigned long long)(m0 + tid);
    }
    rowOff[tid] = tok * (unsigned long long)DIM;
  }
  __syncthreads();

  // rmsnorm: 8 threads per row, 128 elems each
  {
    const int r = tid >> 3;
    const int sub = tid & 7;
    const float* xr = X + rowOff[r];
    float ss = 0.f;
#pragma unroll
    for (int q = 0; q < 32; ++q) {
      float4 v = *(const float4*)(xr + sub * 128 + q * 4);
      ss += v.x * v.x + v.y * v.y + v.z * v.z + v.w * v.w;
    }
    rr[r][sub] = ss;
    __syncthreads();
    if (sub == 0) {
      float t = 0.f;
#pragma unroll
      for (int q = 0; q < 8; ++q) t += rr[r][q];
      rr[r][8] = 32.0f / fmaxf(sqrtf(t), 1e-12f);   // sqrt(1024) = 32
    }
    __syncthreads();
    const float scale = rr[r][8];
#pragma unroll
    for (int q = 0; q < 32; ++q) {
      int k = sub * 128 + q * 4;
      float4 v = *(const float4*)(xr + k);
      XnU[r * XST + (k >> 1)] =
          pack2_bf16(v.x * scale * gamma[k], v.y * scale * gamma[k + 1]);
      XnU[r * XST + (k >> 1) + 1] =
          pack2_bf16(v.z * scale * gamma[k + 2], v.w * scale * gamma[k + 3]);
    }
  }
  __syncthreads();

  v8f zero = {};
  v8f oacc[16];
#pragma unroll
  for (int t = 0; t < 16; ++t) oacc[t] = zero;

  for (int hc = 0; hc < NCHUNK; ++hc) {
    // ---- GEMM1: Hc = gelu(Xn @ W1_chunk^T + b1) ----
    // 16 N-tiles x 2 M-tiles over 8 waves -> 4 tiles per wave.
    v8f hacc[4];
#pragma unroll
    for (int t = 0; t < 4; ++t) hacc[t] = zero;
#pragma unroll 2
    for (int kk = 0; kk < DIM / 32; ++kk) {
      v16bf A = load_a_lds(XnU, XST, mbase + mlow, kk * 32 + khalf * 16);
#pragma unroll
      for (int t = 0; t < 4; ++t) {
        const int j = hc * CH + (ngrp * 4 + t) * 16 + mlow;
        v16bf B = load_b_bf16(W1 + (size_t)j * DIM + kk * 32 + khalf * 16);
        hacc[t] = __builtin_amdgcn_wmma_f32_16x16x32_bf16(
            false, A, false, B, (short)0, hacc[t], false, false);
      }
    }
#pragma unroll
    for (int t = 0; t < 4; ++t) {
      const int jl = (ngrp * 4 + t) * 16 + mlow;   // col within chunk
      const float bias = B1[hc * CH + jl];
#pragma unroll
      for (int rix = 0; rix < 8; ++rix) {
        float hv = gelu_exact(hacc[t][rix] + bias);
        Hcf[(mbase + rix + khalf * 8) * HST + jl] = (__bf16)hv;
      }
    }
    __syncthreads();
    // ---- GEMM2 partial: oacc += Hc @ W2_chunk^T ----
    // 64 N-tiles x 2 M-tiles over 8 waves -> 16 tiles per wave.
#pragma unroll 2
    for (int kk = 0; kk < CH / 32; ++kk) {
      v16bf A = load_a_lds((const unsigned*)Hcf, HST / 2, mbase + mlow,
                           kk * 32 + khalf * 16);
#pragma unroll
      for (int t = 0; t < 16; ++t) {
        const int jcol = ngrp * 256 + t * 16 + mlow;
        v16bf B = load_b_bf16(W2 + (size_t)jcol * DH + hc * CH + kk * 32 +
                              khalf * 16);
        oacc[t] = __builtin_amdgcn_wmma_f32_16x16x32_bf16(
            false, A, false, B, (short)0, oacc[t], false, false);
      }
    }
    __syncthreads();
  }

  // epilogue: + b2, write (light) or scatter-add (heavy)
#pragma unroll
  for (int t = 0; t < 16; ++t) {
    const int col = ngrp * 256 + t * 16 + mlow;
    const float bias = B2[col];
#pragma unroll
    for (int rix = 0; rix < 8; ++rix) {
      const int m = mbase + rix + khalf * 8;
      float v = oacc[t][rix] + bias;
      float* op = out + rowOff[m] + col;
      if (HEAVY) *op += v; else *op = v;
    }
  }
}

// ---------------- launcher ----------------
extern "C" void kernel_launch(void* const* d_in, const int* in_sizes, int n_in,
                              void* d_out, int out_size, void* d_ws,
                              size_t ws_size, hipStream_t stream) {
  const float* x       = (const float*)d_in[0];
  const float* rtok    = (const float*)d_in[1];
  const float* gamma_l = (const float*)d_in[2];
  const float* w1_l    = (const float*)d_in[3];
  const float* b1_l    = (const float*)d_in[4];
  const float* w2_l    = (const float*)d_in[5];
  const float* b2_l    = (const float*)d_in[6];
  const float* gamma_h = (const float*)d_in[7];
  const float* w1_h    = (const float*)d_in[8];
  const float* b1_h    = (const float*)d_in[9];
  const float* w2_h    = (const float*)d_in[10];
  const float* b2_h    = (const float*)d_in[11];
  float* out = (float*)d_out;

  float* s_buf  = (float*)d_ws;            // NTOK floats
  float* sc_buf = s_buf + NTOK;            // NTOK floats
  int*   sel    = (int*)(sc_buf + NTOK);   // NB*KSEL ints
  unsigned* wbase = (unsigned*)(sel + NB * KSEL);
  const int n1l = 512 * 1024, n2l = 1024 * 512;
  const int n1h = 4096 * 1024, n2h = 1024 * 4096;
  __bf16* w1l = (__bf16*)wbase;
  __bf16* w2l = w1l + n1l;
  __bf16* w1h = w2l + n2l;
  __bf16* w2h = w1h + n1h;

  // 0) pre-convert weights to bf16 (read from L2 by all GEMM blocks)
  convert_kernel<<<n1l / (256 * 8), 256, 0, stream>>>(w1_l, (unsigned*)w1l, n1l);
  convert_kernel<<<n2l / (256 * 8), 256, 0, stream>>>(w2_l, (unsigned*)w2l, n2l);
  convert_kernel<<<n1h / (256 * 8), 256, 0, stream>>>(w1_h, (unsigned*)w1h, n1h);
  convert_kernel<<<n2h / (256 * 8), 256, 0, stream>>>(w2_h, (unsigned*)w2h, n2h);
  // 1) router logits
  router_kernel<<<NTOK / 8, 256, 0, stream>>>(x, rtok, s_buf);
  // 2) coor_descent scores
  coor_kernel<<<NB, 1024, 0, stream>>>(s_buf, sc_buf);
  // 3) stable top-k -> selected indices
  rank_kernel<<<NB * (NSEQ / 256), 256, 0, stream>>>(sc_buf, sel);
  // 4) light FFN on all tokens (writes out)
  ffn_kernel<512, 2, false><<<NTOK / TM, 256, 0, stream>>>(
      x, gamma_l, w1l, b1_l, w2l, b2_l, nullptr, out);
  // 5) heavy FFN on routed tokens (scatter-add; straight-through scale == 1)
  ffn_kernel<4096, 16, true><<<(NB * KSEL) / TM, 256, 0, stream>>>(
      x, gamma_h, w1h, b1_h, w2h, b2_h, sel, out);
}
